// Decode_78572131713671
// MI455X (gfx1250) — matrix operand order repaired and checked by reading, not compiled, not run on hardware
//
#include <hip/hip_runtime.h>
#include <hip/hip_bf16.h>

// Decode pipeline for MI455X (gfx1250, wave32, WMMA f16 16x16x32, f32 accum).
// B=64, T=512, E=1024, H=512, LH=512, IN_D=1152, S=66.

#define Bsz 64
#define Tn  512
#define Ed  1024
#define Hd  512
#define LHd 512
#define IND 1152
#define Sd  66
#define G4H 2048
#define NPAD 80
#define KCMB 1536
#define Mtot (Bsz * Tn)   // 32768

typedef __attribute__((ext_vector_type(16))) _Float16 v16h;
typedef __attribute__((ext_vector_type(8)))  _Float16 v8h;
typedef __attribute__((ext_vector_type(8)))  float    v8f;

__device__ __forceinline__ v8f zero8() {
  v8f z = {0.f, 0.f, 0.f, 0.f, 0.f, 0.f, 0.f, 0.f};
  return z;
}

__device__ __forceinline__ float sigf(float x) { return 1.f / (1.f + __expf(-x)); }

__device__ __forceinline__ v8f wmma_f16(v16h a, v16h b, v8f c) {
  // (neg_a, A, neg_b, B, c_mod, C, reuse_a, reuse_b)
  return __builtin_amdgcn_wmma_f32_16x16x32_f16(false, a, false, b, (short)0, c,
                                                false, false);
}

// A fragment: 16x32 f16 tile of row-major X (leading dim ld).
// ISA layout (CDNA5 ISA 7.12.2): lane L -> row M = L%16; with g = L/16:
//   halves 0..7  hold K = k0 + 8g + h
//   halves 8..15 hold K = k0 + 16 + 8g + (h-8)
__device__ __forceinline__ v16h loadA(const _Float16* __restrict__ X, int ld,
                                      int row, int k0, int g) {
  const _Float16* p = X + (size_t)row * ld + k0 + 8 * g;
  v8h lo = *(const v8h*)p;
  v8h hi = *(const v8h*)(p + 16);
  v16h a;
#pragma unroll
  for (int i = 0; i < 8; ++i) { a[i] = lo[i]; a[8 + i] = hi[i]; }
  return a;
}

// B fragment: 32x16 (KxN) f16, sourced from row-major weight W[N,K] (out = X @ W^T).
// Lane L -> column N = L%16; with g = L/16: halves 0..7 hold K = k0 + 8g + h,
// halves 8..15 hold K = k0 + 16 + 8g + (h-8).
__device__ __forceinline__ v16h loadB(const _Float16* __restrict__ W, int ld,
                                      int col, int k0, int g) {
  const _Float16* p = W + (size_t)col * ld + k0 + 8 * g;
  v8h lo = *(const v8h*)p;
  v8h hi = *(const v8h*)(p + 16);
  v16h b;
#pragma unroll
  for (int i = 0; i < 8; ++i) { b[i] = lo[i]; b[8 + i] = hi[i]; }
  return b;
}

// ---------------------------------------------------------------- prep -----
__global__ __launch_bounds__(256) void k_prep(
    const float* __restrict__ fcw, const float* __restrict__ wih,
    const float* __restrict__ whh, const float* __restrict__ cw,
    const float* __restrict__ bih, const float* __restrict__ bhh,
    _Float16* __restrict__ fcw16, _Float16* __restrict__ wih16,
    _Float16* __restrict__ whh16, _Float16* __restrict__ cw16,
    float* __restrict__ gb, _Float16* __restrict__ h0, float* __restrict__ c0) {
  size_t gid = (size_t)blockIdx.x * blockDim.x + threadIdx.x;
  size_t gs = (size_t)gridDim.x * blockDim.x;
  for (size_t i = gid; i < (size_t)LHd * IND; i += gs) fcw16[i] = (_Float16)fcw[i];
  for (size_t i = gid; i < (size_t)G4H * LHd; i += gs) {
    wih16[i] = (_Float16)wih[i];
    whh16[i] = (_Float16)whh[i];
  }
  for (size_t i = gid; i < (size_t)NPAD * KCMB; i += gs) {
    size_t r = i / KCMB;
    cw16[i] = (r < Sd) ? (_Float16)cw[i] : (_Float16)0.f;   // pad rows 66..79 with 0
  }
  for (size_t i = gid; i < G4H; i += gs) gb[i] = bih[i] + bhh[i];
  for (size_t i = gid; i < (size_t)Bsz * Hd; i += gs) {
    h0[i] = (_Float16)0.f;
    c0[i] = 0.f;
  }
}

// ---------------------------------------------------- build z_in + enc16 ---
// One block per (b,t). Window mean over <=7 rows, embedding gathers, f16 pack.
__global__ __launch_bounds__(256) void k_build_zin(
    const float* __restrict__ enc, const float* __restrict__ wlW,
    const float* __restrict__ posW, const int* __restrict__ pid,
    const int* __restrict__ wlid, const int* __restrict__ wlval,
    _Float16* __restrict__ zin, _Float16* __restrict__ V) {
  const int row = blockIdx.x;           // b*T + t
  const int b = row / Tn, t = row % Tn;
  const int tid = threadIdx.x;
  const int wl = wlval[row];
  int start = t - wl;
  if (start < 0) start = 0;
  const int num = t - start;
  const float inv = 1.f / (float)(num > 0 ? num : 1);

  const int c0 = tid * 4;               // 0..1020 (E = 1024)
  float4 s = make_float4(0.f, 0.f, 0.f, 0.f);
  for (int r = start; r < t; ++r) {
    const float4 e = *(const float4*)(enc + ((size_t)b * Tn + r) * Ed + c0);
    s.x += e.x; s.y += e.y; s.z += e.z; s.w += e.w;
  }
  _Float16* zp = zin + (size_t)row * IND;
  zp[c0 + 0] = (_Float16)(s.x * inv);
  zp[c0 + 1] = (_Float16)(s.y * inv);
  zp[c0 + 2] = (_Float16)(s.z * inv);
  zp[c0 + 3] = (_Float16)(s.w * inv);

  // encode row t itself -> comb input cols [H, H+E)
  const float4 et = *(const float4*)(enc + (size_t)row * Ed + c0);
  _Float16* vp = V + (size_t)row * KCMB + Hd;
  vp[c0 + 0] = (_Float16)et.x;
  vp[c0 + 1] = (_Float16)et.y;
  vp[c0 + 2] = (_Float16)et.z;
  vp[c0 + 3] = (_Float16)et.w;

  if (tid < 64) {
    zp[Ed + tid] = (_Float16)wlW[wlid[row] * 64 + tid];
  } else if (tid < 128) {
    const int j = tid - 64;
    zp[Ed + 64 + j] = (_Float16)posW[pid[row] * 64 + j];
  }
}

// ------------------------------------------------ fc GEMM: z = tanh(.) -----
// M=32768, K=1152, N=512. Block tile 128x128 (8 waves, each 32x64).
__global__ __launch_bounds__(256) void k_fc(const _Float16* __restrict__ A,
                                            const _Float16* __restrict__ Bw,
                                            const float* __restrict__ bias,
                                            _Float16* __restrict__ Z) {
  const int lane = threadIdx.x & 31, w = threadIdx.x >> 5;
  const int g = lane >> 4, ln = lane & 15;
  const int m0 = blockIdx.x * 128 + (w >> 1) * 32;
  const int n0 = blockIdx.y * 128 + (w & 1) * 64;
  v8f acc[2][4];
#pragma unroll
  for (int mi = 0; mi < 2; ++mi)
#pragma unroll
    for (int j = 0; j < 4; ++j) acc[mi][j] = zero8();

  for (int k0 = 0; k0 < IND; k0 += 32) {
    v16h a0 = loadA(A, IND, m0 + ln, k0, g);
    v16h a1 = loadA(A, IND, m0 + 16 + ln, k0, g);
#pragma unroll
    for (int j = 0; j < 4; ++j) {
      v16h b = loadB(Bw, IND, n0 + j * 16 + ln, k0, g);
      acc[0][j] = wmma_f16(a0, b, acc[0][j]);
      acc[1][j] = wmma_f16(a1, b, acc[1][j]);
    }
  }
#pragma unroll
  for (int mi = 0; mi < 2; ++mi)
#pragma unroll
    for (int j = 0; j < 4; ++j) {
      const int n = n0 + j * 16 + ln;
      const float bn = bias[n];
#pragma unroll
      for (int v = 0; v < 8; ++v) {
        const int m = m0 + mi * 16 + 8 * g + v;
        float val = tanhf(acc[mi][j][v] + bn);
        if ((m % Tn) == 0) val = 0.f;   // z_0 stays zero
        Z[(size_t)m * LHd + n] = (_Float16)val;
      }
    }
}

// ---------------------------------- gates_x GEMM: gx = z @ w_ih^T + b ------
// M=32768, K=512, N=2048, fp32 output.
__global__ __launch_bounds__(256) void k_gx(const _Float16* __restrict__ A,
                                            const _Float16* __restrict__ Bw,
                                            const float* __restrict__ gb,
                                            float* __restrict__ GX) {
  const int lane = threadIdx.x & 31, w = threadIdx.x >> 5;
  const int g = lane >> 4, ln = lane & 15;
  const int m0 = blockIdx.x * 128 + (w >> 1) * 32;
  const int n0 = blockIdx.y * 128 + (w & 1) * 64;
  v8f acc[2][4];
#pragma unroll
  for (int mi = 0; mi < 2; ++mi)
#pragma unroll
    for (int j = 0; j < 4; ++j) acc[mi][j] = zero8();

  for (int k0 = 0; k0 < LHd; k0 += 32) {
    v16h a0 = loadA(A, LHd, m0 + ln, k0, g);
    v16h a1 = loadA(A, LHd, m0 + 16 + ln, k0, g);
#pragma unroll
    for (int j = 0; j < 4; ++j) {
      v16h b = loadB(Bw, LHd, n0 + j * 16 + ln, k0, g);
      acc[0][j] = wmma_f16(a0, b, acc[0][j]);
      acc[1][j] = wmma_f16(a1, b, acc[1][j]);
    }
  }
#pragma unroll
  for (int mi = 0; mi < 2; ++mi)
#pragma unroll
    for (int j = 0; j < 4; ++j) {
      const int n = n0 + j * 16 + ln;
      const float bn = gb[n];
#pragma unroll
      for (int v = 0; v < 8; ++v) {
        const int m = m0 + mi * 16 + 8 * g + v;
        GX[(size_t)m * G4H + n] = acc[mi][j][v] + bn;
      }
    }
}

// ------------------------------------------------ one LSTM recurrence step -
// Block owns 32 hidden columns; 8 waves = 2 row-halves x 4 gates.
// gates = gx[:,t,:] + h_prev @ w_hh^T ; fused cell update.
__global__ __launch_bounds__(256) void k_step(
    const _Float16* __restrict__ hp, _Float16* __restrict__ hn,
    const _Float16* __restrict__ whh, const float* __restrict__ gx,
    float* __restrict__ cst, _Float16* __restrict__ V, int t) {
  __shared__ float lg[4][64][32];       // 32 KB
  const int lane = threadIdx.x & 31, w = threadIdx.x >> 5;
  const int g = lane >> 4, ln = lane & 15;
  const int n0 = blockIdx.x * 32;       // hidden-column block
  const int gate = w & 3;
  const int mb = (w >> 2) * 32;         // row base (B = 64 rows)
  v8f acc[2][2];
#pragma unroll
  for (int mi = 0; mi < 2; ++mi)
#pragma unroll
    for (int j = 0; j < 2; ++j) acc[mi][j] = zero8();

  for (int k0 = 0; k0 < Hd; k0 += 32) {
    v16h a0 = loadA(hp, Hd, mb + ln, k0, g);
    v16h a1 = loadA(hp, Hd, mb + 16 + ln, k0, g);
#pragma unroll
    for (int j = 0; j < 2; ++j) {
      v16h b = loadB(whh, Hd, gate * Hd + n0 + j * 16 + ln, k0, g);
      acc[0][j] = wmma_f16(a0, b, acc[0][j]);
      acc[1][j] = wmma_f16(a1, b, acc[1][j]);
    }
  }
#pragma unroll
  for (int mi = 0; mi < 2; ++mi)
#pragma unroll
    for (int j = 0; j < 2; ++j) {
      const int c = j * 16 + ln;
#pragma unroll
      for (int v = 0; v < 8; ++v) lg[gate][mb + mi * 16 + 8 * g + v][c] = acc[mi][j][v];
    }
  __syncthreads();

  for (int e = threadIdx.x; e < 64 * 32; e += 256) {
    const int m = e >> 5, c = e & 31;
    const int col = n0 + c;
    const size_t grow = ((size_t)m * Tn + t) * G4H;
    const float pi = lg[0][m][c] + gx[grow + col];
    const float pf = lg[1][m][c] + gx[grow + Hd + col];
    const float pg = lg[2][m][c] + gx[grow + 2 * Hd + col];
    const float po = lg[3][m][c] + gx[grow + 3 * Hd + col];
    const float cv = cst[m * Hd + col];
    const float cn = sigf(pf) * cv + sigf(pi) * tanhf(pg);
    const float hv = sigf(po) * tanhf(cn);
    cst[m * Hd + col] = cn;
    hn[m * Hd + col] = (_Float16)hv;
    V[((size_t)m * Tn + t) * KCMB + col] = (_Float16)hv;   // comb input cols [0,H)
  }
}

// ------------------------------------- comb GEMM + mask/APP epilogue -------
// M=32768, K=1536, N=80 (padded from 66). Block 128 rows, wave = 16 rows x 80.
__global__ __launch_bounds__(256) void k_comb(const _Float16* __restrict__ A,
                                              const _Float16* __restrict__ Bw,
                                              const float* __restrict__ cb,
                                              const int* __restrict__ lens,
                                              float* __restrict__ out) {
  const int lane = threadIdx.x & 31, w = threadIdx.x >> 5;
  const int g = lane >> 4, ln = lane & 15;
  const int m0 = blockIdx.x * 128 + w * 16;
  v8f acc[5];
#pragma unroll
  for (int j = 0; j < 5; ++j) acc[j] = zero8();

  for (int k0 = 0; k0 < KCMB; k0 += 32) {
    v16h a = loadA(A, KCMB, m0 + ln, k0, g);
#pragma unroll
    for (int j = 0; j < 5; ++j) {
      v16h b = loadB(Bw, KCMB, j * 16 + ln, k0, g);
      acc[j] = wmma_f16(a, b, acc[j]);
    }
  }
#pragma unroll
  for (int j = 0; j < 5; ++j) {
    const int n = j * 16 + ln;
    if (n < Sd) {
      const float bn = cb[n];
#pragma unroll
      for (int v = 0; v < 8; ++v) {
        const int m = m0 + 8 * g + v;
        const int b = m / Tn, t = m % Tn;
        float val = acc[j][v] + bn;
        if (t == 0 && n == 0) val = -1e30f;     // APP forbidden at first char
        if (t >= lens[b]) val = 0.f;            // padded steps emit zero rows
        out[(size_t)m * Sd + n] = val;
      }
    }
  }
}

// --------------------------------------------------------------- launch ----
extern "C" void kernel_launch(void* const* d_in, const int* in_sizes, int n_in,
                              void* d_out, int out_size, void* d_ws, size_t ws_size,
                              hipStream_t stream) {
  (void)in_sizes; (void)n_in; (void)out_size; (void)ws_size;
  const float* enc  = (const float*)d_in[0];
  const float* posW = (const float*)d_in[1];
  const float* wlW  = (const float*)d_in[2];
  const float* fcw  = (const float*)d_in[3];
  const float* fcb  = (const float*)d_in[4];
  const float* wih  = (const float*)d_in[5];
  const float* whh  = (const float*)d_in[6];
  const float* bih  = (const float*)d_in[7];
  const float* bhh  = (const float*)d_in[8];
  const float* cw   = (const float*)d_in[9];
  const float* cb   = (const float*)d_in[10];
  const int* pid    = (const int*)d_in[11];
  const int* wlid   = (const int*)d_in[12];
  const int* wlval  = (const int*)d_in[13];
  const int* lens   = (const int*)d_in[14];
  float* out = (float*)d_out;

  char* ws = (char*)d_ws;
  size_t off = 0;
  auto take = [&](size_t bytes) -> char* {
    char* p = ws + off;
    off += (bytes + 255) & ~(size_t)255;
    return p;
  };
  _Float16* zin   = (_Float16*)take((size_t)Mtot * IND * 2);   //  75.5 MB
  _Float16* V     = (_Float16*)take((size_t)Mtot * KCMB * 2);  // 100.7 MB
  _Float16* z16   = (_Float16*)take((size_t)Mtot * LHd * 2);   //  33.6 MB
  float*    gx    = (float*)   take((size_t)Mtot * G4H * 4);   // 268.4 MB
  _Float16* fcw16 = (_Float16*)take((size_t)LHd * IND * 2);
  _Float16* wih16 = (_Float16*)take((size_t)G4H * LHd * 2);
  _Float16* whh16 = (_Float16*)take((size_t)G4H * Hd * 2);
  _Float16* cw16  = (_Float16*)take((size_t)NPAD * KCMB * 2);
  float*    gb    = (float*)   take((size_t)G4H * 4);
  _Float16* hA    = (_Float16*)take((size_t)Bsz * Hd * 2);
  _Float16* hB    = (_Float16*)take((size_t)Bsz * Hd * 2);
  float*    cst   = (float*)   take((size_t)Bsz * Hd * 4);

  k_prep<<<2048, 256, 0, stream>>>(fcw, wih, whh, cw, bih, bhh,
                                   fcw16, wih16, whh16, cw16, gb, hA, cst);
  k_build_zin<<<Mtot, 256, 0, stream>>>(enc, wlW, posW, pid, wlid, wlval, zin, V);
  k_fc<<<dim3(Mtot / 128, LHd / 128), 256, 0, stream>>>(zin, fcw16, fcb, z16);
  k_gx<<<dim3(Mtot / 128, G4H / 128), 256, 0, stream>>>(z16, wih16, gb, gx);
  for (int t = 0; t < Tn; ++t) {
    const _Float16* hp = (t & 1) ? hB : hA;
    _Float16* hn = (t & 1) ? hA : hB;
    k_step<<<Hd / 32, 256, 0, stream>>>(hp, hn, whh16, gx, cst, V, t);
  }
  k_comb<<<Mtot / 128, 256, 0, stream>>>(V, cw16, cb, lens, out);
}
